// PrunableAttention_21947282883258
// MI455X (gfx1250) — compile-verified, matrix operand order checked by reading, not run on hardware
//
#include <hip/hip_runtime.h>
#include <hip/hip_bf16.h>
#include <math.h>

// MHA block: B=2, N=2048, C=1024, H=16, D=64.
// Strategy: bf16 V_WMMA_F32_16X16X32_BF16 for all three GEMM stages,
// flash-attention (no N^2 materialization), fp32 accumulation throughout.

typedef __bf16 bf16_t;
typedef bf16_t v16bf __attribute__((ext_vector_type(16)));
typedef float  v8f   __attribute__((ext_vector_type(8)));
typedef unsigned short u16;
typedef u16 v8u16  __attribute__((ext_vector_type(8)));
typedef u16 v16u16 __attribute__((ext_vector_type(16)));

#define B_ 2
#define N_ 2048
#define C_ 1024
#define H_ 16
#define D_ 64

__device__ __forceinline__ u16 f2bf(float f) {           // RNE f32 -> bf16
  unsigned u = __float_as_uint(f);
  u += 0x7fffu + ((u >> 16) & 1u);
  return (u16)(u >> 16);
}

__device__ __forceinline__ v8f wmma_bf16(v16bf a, v16bf b, v8f c) {
  return __builtin_amdgcn_wmma_f32_16x16x32_bf16(false, a, false, b,
                                                 (short)0, c, false, false);
}

// NT fragment loader: works for both A (16xK, M rows) and B (Kx16 from a
// row-major [N,K] operand). Lane L -> row base+(L%16); K chunks
// [k0+8*(L/16) .. +7] and [+16 .. +23]. Two 16B global loads.
__device__ __forceinline__ v16bf load_frag_nt(const u16* __restrict__ p,
                                              int ld, int row, int k0, int lane) {
  int r  = row + (lane & 15);
  int kc = k0 + ((lane >> 4) << 3);
  const u16* q = p + r * ld + kc;
  v8u16 lo = *(const v8u16*)q;
  v8u16 hi = *(const v8u16*)(q + 16);
  v16u16 f;
#pragma unroll
  for (int i = 0; i < 8; ++i) { f[i] = lo[i]; f[i + 8] = hi[i]; }
  return __builtin_bit_cast(v16bf, f);
}

// ---------------------------------------------------------------- cvt f32->bf16
__global__ void cvt_f32_bf16(const float* __restrict__ in, u16* __restrict__ out,
                             int n) {
  int i = blockIdx.x * blockDim.x + threadIdx.x;
  int stride = gridDim.x * blockDim.x;
  for (; i < n; i += stride) out[i] = f2bf(in[i]);
}

// ---------------------------------------------------------------- QKV GEMM
// y[m,o] = sum_k x[m,k]*w_qkv[o,k] + b[o];  M=4096, O=3072, K=1024.
// 256 thr = 8 waves (2x4), 32x32 tile/wave -> 64x128 block tile.
// Epilogue scatters to Q[B,H,N,D], K[B,H,N,D], Vt[B,H,D,N] (bf16).
__global__ __launch_bounds__(256)
void qkv_gemm(const u16* __restrict__ xb, const u16* __restrict__ wb,
              const float* __restrict__ bias,
              u16* __restrict__ Qb, u16* __restrict__ Kb, u16* __restrict__ Vt) {
  const int lane = threadIdx.x & 31;
  const int wid  = threadIdx.x >> 5;
  const int m0 = blockIdx.y * 64  + (wid >> 2) * 32;
  const int o0 = blockIdx.x * 128 + (wid & 3) * 32;

  v8f acc[2][2] = {};
  for (int k = 0; k < C_; k += 32) {
    v16bf a0 = load_frag_nt(xb, C_, m0,      k, lane);
    v16bf a1 = load_frag_nt(xb, C_, m0 + 16, k, lane);
    v16bf b0 = load_frag_nt(wb, C_, o0,      k, lane);
    v16bf b1 = load_frag_nt(wb, C_, o0 + 16, k, lane);
    acc[0][0] = wmma_bf16(a0, b0, acc[0][0]);
    acc[0][1] = wmma_bf16(a0, b1, acc[0][1]);
    acc[1][0] = wmma_bf16(a1, b0, acc[1][0]);
    acc[1][1] = wmma_bf16(a1, b1, acc[1][1]);
  }

  const int lg = lane >> 4;
#pragma unroll
  for (int mi = 0; mi < 2; ++mi)
#pragma unroll
    for (int ni = 0; ni < 2; ++ni) {
      int o = o0 + ni * 16 + (lane & 15);
      float bv = bias[o];
      int t = o >> 10;          // 0=q 1=k 2=v
      int h = (o >> 6) & 15;
      int d = o & 63;
#pragma unroll
      for (int i = 0; i < 8; ++i) {
        int m  = m0 + mi * 16 + i + lg * 8;
        int bb = m >> 11;       // batch
        int n  = m & (N_ - 1);
        u16 val = f2bf(acc[mi][ni][i] + bv);
        int bh = bb * H_ + h;
        if (t == 0)      Qb[(bh * N_ + n) * D_ + d] = val;
        else if (t == 1) Kb[(bh * N_ + n) * D_ + d] = val;
        else             Vt[(bh * D_ + d) * N_ + n] = val;   // V transposed
      }
    }
}

// ---------------------------------------------------------------- flash attention
// 128 thr = 4 waves; wave = 16 query rows; block = 64 rows of one (b,h).
// Per 32-key step: 4 WMMA for S = Q*K^T (D=64 -> two K=32 steps x 2 n-tiles),
// online softmax in C-fragment layout (row stats live per 16-lane half),
// P bounced through wave-private LDS into A layout, 4 WMMA for P*V.
__global__ __launch_bounds__(128)
void flash_attn(const u16* __restrict__ Qb, const u16* __restrict__ Kb,
                const u16* __restrict__ Vt, u16* __restrict__ Ab) {
  __shared__ u16 plds[4 * 16 * 32];            // 4 KB: 16x32 bf16 tile per wave
  const int lane = threadIdx.x & 31;
  const int wid  = threadIdx.x >> 5;
  const int nq = N_ / 64;
  const int qt = blockIdx.x % nq;
  const int bh = blockIdx.x / nq;
  const int b = bh >> 4, h = bh & 15;
  const int q0 = qt * 64 + wid * 16;
  const int lg = lane >> 4;

  const u16* Qp = Qb + (bh * N_ + q0) * D_;
  const u16* Kp = Kb + bh * N_ * D_;
  const u16* Vp = Vt + bh * D_ * N_;

  v16bf qa0 = load_frag_nt(Qp, D_, 0, 0,  lane);
  v16bf qa1 = load_frag_nt(Qp, D_, 0, 32, lane);

  float mrow[8], lrow[8];
  v8f oacc[4] = {};
#pragma unroll
  for (int i = 0; i < 8; ++i) { mrow[i] = -1e30f; lrow[i] = 0.0f; }

  const float scale = 0.125f;                  // D^-0.5
  const int base = wid * (16 * 32);

  for (int j = 0; j < N_; j += 32) {
    // ---- S = Q * K^T  (16 queries x 32 keys)
    v8f s0 = {}, s1 = {};
    v16bf kb0 = load_frag_nt(Kp, D_, j,      0,  lane);
    v16bf kb1 = load_frag_nt(Kp, D_, j,      32, lane);
    v16bf kb2 = load_frag_nt(Kp, D_, j + 16, 0,  lane);
    v16bf kb3 = load_frag_nt(Kp, D_, j + 16, 32, lane);
    s0 = wmma_bf16(qa0, kb0, s0);  s0 = wmma_bf16(qa1, kb1, s0);
    s1 = wmma_bf16(qa0, kb2, s1);  s1 = wmma_bf16(qa1, kb3, s1);

    // ---- online softmax: rows split per 16-lane half (C-frag layout:
    //      VGPR i, lanes 0-15 -> row i; lanes 16-31 -> row i+8).
#pragma unroll
    for (int i = 0; i < 8; ++i) {
      float a0 = s0[i] * scale, a1 = s1[i] * scale;
      float mnew = fmaxf(a0, a1);
#pragma unroll
      for (int off = 8; off >= 1; off >>= 1)
        mnew = fmaxf(mnew, __shfl_xor(mnew, off));  // stays inside 16-lane half
      mnew = fmaxf(mnew, mrow[i]);
      float alpha = __expf(mrow[i] - mnew);
      mrow[i] = mnew;
      float p0 = __expf(a0 - mnew), p1 = __expf(a1 - mnew);
      float rs = p0 + p1;
#pragma unroll
      for (int off = 8; off >= 1; off >>= 1)
        rs += __shfl_xor(rs, off);
      lrow[i] = lrow[i] * alpha + rs;
      s0[i] = p0; s1[i] = p1;
      oacc[0][i] *= alpha; oacc[1][i] *= alpha;
      oacc[2][i] *= alpha; oacc[3][i] *= alpha;
    }

    // ---- P: C-layout -> LDS (row-major 16x32 bf16) -> A-layout reload.
#pragma unroll
    for (int i = 0; i < 8; ++i) {
      int r = i + lg * 8;
      plds[base + r * 32 + (lane & 15)]      = f2bf(s0[i]);
      plds[base + r * 32 + 16 + (lane & 15)] = f2bf(s1[i]);
    }
    asm volatile("s_wait_dscnt 0" ::: "memory");   // cross-lane RAW in LDS

    v16bf pa;
    {
      int r  = lane & 15;
      int kc = (lane >> 4) << 3;
      const u16* pp = &plds[base + r * 32 + kc];
      v8u16 lo = *(const v8u16*)pp;
      v8u16 hi = *(const v8u16*)(pp + 16);
      v16u16 f;
#pragma unroll
      for (int i = 0; i < 8; ++i) { f[i] = lo[i]; f[i + 8] = hi[i]; }
      pa = __builtin_bit_cast(v16bf, f);
    }

    // ---- O += P * V  (Vt[D,N] makes the B operand contiguous along keys)
#pragma unroll
    for (int nt = 0; nt < 4; ++nt) {
      v16bf vb = load_frag_nt(Vp, N_, nt * 16, j, lane);
      oacc[nt] = wmma_bf16(pa, vb, oacc[nt]);
    }
  }

  // ---- epilogue: O /= l, write bf16 to Ab[B,N,C] for the proj GEMM
#pragma unroll
  for (int i = 0; i < 8; ++i) {
    float inv = 1.0f / lrow[i];
    int row = q0 + i + lg * 8;
#pragma unroll
    for (int nt = 0; nt < 4; ++nt) {
      int d = nt * 16 + (lane & 15);
      Ab[(b * N_ + row) * C_ + h * D_ + d] = f2bf(oacc[nt][i] * inv);
    }
  }
}

// ---------------------------------------------------------------- proj GEMM
// out[m,o] = sum_c a[m,c]*w_proj[o,c] + b[o];  M=4096, O=1024, K=1024. fp32 out.
__global__ __launch_bounds__(256)
void proj_gemm(const u16* __restrict__ ab, const u16* __restrict__ wb,
               const float* __restrict__ bias, float* __restrict__ out) {
  const int lane = threadIdx.x & 31;
  const int wid  = threadIdx.x >> 5;
  const int m0 = blockIdx.y * 64  + (wid >> 2) * 32;
  const int o0 = blockIdx.x * 128 + (wid & 3) * 32;

  v8f acc[2][2] = {};
  for (int k = 0; k < C_; k += 32) {
    v16bf a0 = load_frag_nt(ab, C_, m0,      k, lane);
    v16bf a1 = load_frag_nt(ab, C_, m0 + 16, k, lane);
    v16bf b0 = load_frag_nt(wb, C_, o0,      k, lane);
    v16bf b1 = load_frag_nt(wb, C_, o0 + 16, k, lane);
    acc[0][0] = wmma_bf16(a0, b0, acc[0][0]);
    acc[0][1] = wmma_bf16(a0, b1, acc[0][1]);
    acc[1][0] = wmma_bf16(a1, b0, acc[1][0]);
    acc[1][1] = wmma_bf16(a1, b1, acc[1][1]);
  }

  const int lg = lane >> 4;
#pragma unroll
  for (int mi = 0; mi < 2; ++mi)
#pragma unroll
    for (int ni = 0; ni < 2; ++ni) {
      int o = o0 + ni * 16 + (lane & 15);
      float bv = bias[o];
#pragma unroll
      for (int i = 0; i < 8; ++i) {
        int m = m0 + mi * 16 + i + lg * 8;
        out[m * C_ + o] = acc[mi][ni][i] + bv;
      }
    }
}

// ---------------------------------------------------------------- launch
extern "C" void kernel_launch(void* const* d_in, const int* in_sizes, int n_in,
                              void* d_out, int out_size, void* d_ws, size_t ws_size,
                              hipStream_t stream) {
  const float* x      = (const float*)d_in[0];
  const float* w_qkv  = (const float*)d_in[1];
  const float* b_qkv  = (const float*)d_in[2];
  const float* w_proj = (const float*)d_in[3];
  const float* b_proj = (const float*)d_in[4];
  float* out = (float*)d_out;

  // workspace layout (bytes), total ~50.3 MB
  char* ws = (char*)d_ws;
  u16* xb     = (u16*)(ws + 0);         //  8,388,608  x bf16 [4096,1024]
  u16* wqkvb  = (u16*)(ws + 8388608);   //  6,291,456  w_qkv bf16 [3072,1024]
  u16* wprojb = (u16*)(ws + 14680064);  //  2,097,152  w_proj bf16 [1024,1024]
  u16* Qb     = (u16*)(ws + 16777216);  //  8,388,608  Q bf16 [B,H,N,D]
  u16* Kb     = (u16*)(ws + 25165824);  //  8,388,608  K bf16 [B,H,N,D]
  u16* Vt     = (u16*)(ws + 33554432);  //  8,388,608  V^T bf16 [B,H,D,N]
  u16* Ab     = (u16*)(ws + 41943040);  //  8,388,608  attn-out bf16 [B,N,C]

  cvt_f32_bf16<<<2048, 256, 0, stream>>>(x,      xb,     B_ * N_ * C_);
  cvt_f32_bf16<<<2048, 256, 0, stream>>>(w_qkv,  wqkvb,  3 * C_ * C_);
  cvt_f32_bf16<<<1024, 256, 0, stream>>>(w_proj, wprojb, C_ * C_);

  qkv_gemm<<<dim3(3 * C_ / 128, (B_ * N_) / 64), 256, 0, stream>>>(
      xb, wqkvb, b_qkv, Qb, Kb, Vt);

  flash_attn<<<B_ * H_ * (N_ / 64), 128, 0, stream>>>(Qb, Kb, Vt, Ab);

  proj_gemm<<<dim3(C_ / 128, (B_ * N_) / 64), 256, 0, stream>>>(
      Ab, wprojb, b_proj, out);
}
